// HGINLayer_80307298500977
// MI455X (gfx1250) — matrix operand order repaired and checked by reading, not compiled, not run on hardware
//
#include <hip/hip_runtime.h>

#define DD 128

typedef __attribute__((ext_vector_type(16))) __bf16          v16bf;
typedef __attribute__((ext_vector_type(16))) unsigned short  v16u;
typedef __attribute__((ext_vector_type(8)))  float           v8f;
typedef __attribute__((ext_vector_type(4)))  int             v4i_;

typedef __attribute__((address_space(1))) v4i_* gv4i_p;   // global int4*
typedef __attribute__((address_space(3))) v4i_* lv4i_p;   // LDS int4*

#if defined(__has_builtin)
#  if __has_builtin(__builtin_amdgcn_global_load_async_to_lds_b128)
#    define USE_ASYNC_LDS 1
#  endif
#endif
#ifndef USE_ASYNC_LDS
#  define USE_ASYNC_LDS 0
#endif

__device__ __forceinline__ unsigned short f32_to_bf16_bits(float f) {
  __bf16 b = (__bf16)f;                       // HW round-to-nearest-even (v_cvt_*_bf16_f32)
  return __builtin_bit_cast(unsigned short, b);
}

// ---------------------------------------------------------------------------
// Pack a 128x128 fp32 row-major weight W[k][n] into WMMA B-fragment order:
// frag index ((kc*8 + ct)*32 + lane), 16 bf16 per lane:
//   lane<16 : k = kc*32 + j       (j=0..15), n = ct*16 + lane
//   lane>=16: k = kc*32 + 16 + j,            n = ct*16 + (lane-16)
// ---------------------------------------------------------------------------
__global__ void pack_weight(const float* __restrict__ W, unsigned short* __restrict__ out) {
  int tid = blockIdx.x * blockDim.x + threadIdx.x;  // 0..16383
  int j  = tid & 15;
  int t  = (tid >> 4) & 31;
  int ct = (tid >> 9) & 7;
  int kc = tid >> 12;
  int k  = kc * 32 + (t & 16) + j;
  int n  = ct * 16 + (t & 15);
  out[tid] = f32_to_bf16_bits(W[k * DD + n]);
}

// ---------------------------------------------------------------------------
// One wave per edge: agg[dst[e]] += feat[src[e]]  (D=128, float4 per lane)
// ---------------------------------------------------------------------------
__global__ void scatter_add(const float* __restrict__ feat, const int* __restrict__ src,
                            const int* __restrict__ dst, float* __restrict__ agg, int E) {
  int e    = (blockIdx.x * blockDim.x + threadIdx.x) >> 5;
  int lane = threadIdx.x & 31;
  if (e >= E) return;
  int s = src[e], d = dst[e];
  float4 v = ((const float4*)(feat + (size_t)s * DD))[lane];
  float* dp = agg + (size_t)d * DD + lane * 4;
  atomicAdd(dp + 0, v.x);
  atomicAdd(dp + 1, v.y);
  atomicAdd(dp + 2, v.z);
  atomicAdd(dp + 3, v.w);
}

// ---------------------------------------------------------------------------
// Fused GEMM: out[m][n] = act( A )[m][:] @ W(bf16, packed, staged in LDS) + bias[n]
//   HAS_A1 : a = (1+eps)*A0 + A1          (GIN combine)
//   HAS_AFF: a = relu(a*colS[k] + colT[k]) (BN folded + ReLU)
//   HAS_STATS: per-column sum/sumsq of output -> stats[0..127]/[128..255]
// Block: 128 rows x 128 cols, 256 threads = 8 waves; wave = 16-row stripe,
// 8 accumulator tiles (8 x v8f), 4 k-chunks x 8 WMMA = 32 v_wmma per wave.
// ---------------------------------------------------------------------------
template <bool HAS_A1, bool HAS_AFF, bool HAS_STATS>
__global__ void __launch_bounds__(256)
gemm128_t(const float* __restrict__ A0, const float* __restrict__ A1,
          const float* __restrict__ epsPtr,
          const float* __restrict__ colS, const float* __restrict__ colT,
          const unsigned short* __restrict__ Wpk, const float* __restrict__ bias,
          float* __restrict__ out, float* __restrict__ stats, int M)
{
  __shared__ unsigned short s_w[DD * DD];  // 32 KB packed bf16 weight
  __shared__ float s_sum[DD];
  __shared__ float s_sq[DD];

  // ---- Stage packed W into LDS once per block (async CDNA5 path) ----------
#if USE_ASYNC_LDS
  {
    unsigned long long g = (unsigned long long)(const void*)Wpk;
    // Flat LDS address low 32 bits == LDS byte offset (ISA aperture rule)
    unsigned int l = (unsigned int)(unsigned long long)(const void*)&s_w[0];
    unsigned int byteOff = threadIdx.x * 16u;
    #pragma unroll
    for (int i = 0; i < 8; ++i) {
      __builtin_amdgcn_global_load_async_to_lds_b128(
          (gv4i_p)(g + byteOff),
          (lv4i_p)(l + byteOff),
          0, 0);
      byteOff += 256u * 16u;
    }
  }
#else
  {
    #pragma unroll
    for (int i = 0; i < 8; ++i) {
      int idx = threadIdx.x + i * 256;  // uint4 index; 2048 total
      ((uint4*)s_w)[idx] = ((const uint4*)Wpk)[idx];
    }
  }
#endif
  if (HAS_STATS) {
    if (threadIdx.x < DD) { s_sum[threadIdx.x] = 0.f; s_sq[threadIdx.x] = 0.f; }
  }
#if USE_ASYNC_LDS
#  if defined(__has_builtin) && __has_builtin(__builtin_amdgcn_s_wait_asynccnt)
  __builtin_amdgcn_s_wait_asynccnt(0);
#  else
  asm volatile("s_wait_asynccnt 0" ::: "memory");
#  endif
#endif
  __syncthreads();

  const int wave = threadIdx.x >> 5;
  const int lane = threadIdx.x & 31;
  const int h    = lane >> 4;      // lane half selects K sub-range
  const int l15  = lane & 15;
  const int m    = blockIdx.x * 128 + wave * 16 + l15;   // A-fragment row
  const bool rowOK = (m < M);
  float scale = 1.0f;
  if (HAS_A1) scale = 1.0f + epsPtr[0];

  const float* a0row = A0 + (size_t)(rowOK ? m : 0) * DD;
  const float* a1row = HAS_A1 ? (A1 + (size_t)(rowOK ? m : 0) * DD) : nullptr;

  v8f acc[8];
  #pragma unroll
  for (int i = 0; i < 8; ++i)
    #pragma unroll
    for (int r = 0; r < 8; ++r) acc[i][r] = 0.0f;

  const v16u* Wfrag = (const v16u*)s_w;   // LDS-resident fragments

  #pragma unroll
  for (int kc = 0; kc < 4; ++kc) {
    const int k0  = kc * 32;
    const int kb0 = k0 + 16 * h;       // first 8 of 16 contiguous K for this lane half
    const int kb1 = k0 + 16 * h + 8;   // second 8
    float va[16];
    #pragma unroll
    for (int i = 0; i < 8; ++i) { va[i] = a0row[kb0 + i]; va[8 + i] = a0row[kb1 + i]; }
    if (HAS_A1) {
      #pragma unroll
      for (int i = 0; i < 8; ++i) {
        va[i]     = scale * va[i]     + a1row[kb0 + i];
        va[8 + i] = scale * va[8 + i] + a1row[kb1 + i];
      }
    }
    if (HAS_AFF) {
      #pragma unroll
      for (int i = 0; i < 8; ++i) {
        va[i]     = fmaxf(va[i]     * colS[kb0 + i] + colT[kb0 + i], 0.0f);
        va[8 + i] = fmaxf(va[8 + i] * colS[kb1 + i] + colT[kb1 + i], 0.0f);
      }
    }
    if (!rowOK) {
      #pragma unroll
      for (int i = 0; i < 16; ++i) va[i] = 0.0f;
    }
    v16bf a;
    #pragma unroll
    for (int i = 0; i < 16; ++i) a[i] = (__bf16)va[i];

    #pragma unroll
    for (int ct = 0; ct < 8; ++ct) {
      v16u  bu = Wfrag[(kc * 8 + ct) * 32 + lane];   // 2x ds_load_b128
      v16bf b  = __builtin_bit_cast(v16bf, bu);
      acc[ct] = __builtin_amdgcn_wmma_f32_16x16x32_bf16(
          false, a, false, b, (short)0, acc[ct], false, false);
    }
  }

  // Epilogue: C/D layout: VGPR r -> row (wave*16 + r + 8*h), col = ct*16 + l15
  const int crow = blockIdx.x * 128 + wave * 16 + 8 * h;
  #pragma unroll
  for (int ct = 0; ct < 8; ++ct) {
    const int n  = ct * 16 + l15;
    const float bn = bias[n];
    float lsum = 0.f, lsq = 0.f;
    #pragma unroll
    for (int r = 0; r < 8; ++r) {
      int row = crow + r;
      if (row < M) {
        float v = acc[ct][r] + bn;
        out[(size_t)row * DD + n] = v;
        lsum += v; lsq += v * v;
      }
    }
    if (HAS_STATS) { atomicAdd(&s_sum[n], lsum); atomicAdd(&s_sq[n], lsq); }
  }
  if (HAS_STATS) {
    __syncthreads();
    if (threadIdx.x < DD) {
      atomicAdd(&stats[threadIdx.x],      s_sum[threadIdx.x]);
      atomicAdd(&stats[DD + threadIdx.x], s_sq[threadIdx.x]);
    }
  }
}

// ---------------------------------------------------------------------------
// BN: fold batch-stats + affine into per-column scale/shift:
//   s = rsqrt(var+1e-5)*g,  t = beta - mu*s
// ---------------------------------------------------------------------------
__global__ void bn_make(const float* __restrict__ stats, const float* __restrict__ g,
                        const float* __restrict__ beta, float invN,
                        float* __restrict__ bnS, float* __restrict__ bnT) {
  int n = threadIdx.x;
  float mu  = stats[n] * invN;
  float var = stats[DD + n] * invN - mu * mu;
  float s   = rsqrtf(var + 1e-5f) * g[n];
  bnS[n] = s;
  bnT[n] = beta[n] - mu * s;
}

// ---------------------------------------------------------------------------
extern "C" void kernel_launch(void* const* d_in, const int* in_sizes, int n_in,
                              void* d_out, int out_size, void* d_ws, size_t ws_size,
                              hipStream_t stream) {
  const float* hT     = (const float*)d_in[0];
  const float* hA     = (const float*)d_in[1];
  const float* hB     = (const float*)d_in[2];
  const int* src_ta   = (const int*)d_in[3];
  const int* dst_ta   = (const int*)d_in[4];
  const int* src_tb   = (const int*)d_in[5];
  const int* dst_tb   = (const int*)d_in[6];
  const int* src_at   = (const int*)d_in[7];
  const int* dst_at   = (const int*)d_in[8];
  const int* src_bt   = (const int*)d_in[9];
  const int* dst_bt   = (const int*)d_in[10];
  const float* eps_ta = (const float*)d_in[11];
  const float* eps_tb = (const float*)d_in[12];
  const float* eps_rt = (const float*)d_in[13];
  const float* ta_W1 = (const float*)d_in[14]; const float* ta_b1 = (const float*)d_in[15];
  const float* ta_g  = (const float*)d_in[16]; const float* ta_be = (const float*)d_in[17];
  const float* ta_W2 = (const float*)d_in[18]; const float* ta_b2 = (const float*)d_in[19];
  const float* tb_W1 = (const float*)d_in[20]; const float* tb_b1 = (const float*)d_in[21];
  const float* tb_g  = (const float*)d_in[22]; const float* tb_be = (const float*)d_in[23];
  const float* tb_W2 = (const float*)d_in[24]; const float* tb_b2 = (const float*)d_in[25];
  const float* rt_W1 = (const float*)d_in[26]; const float* rt_b1 = (const float*)d_in[27];
  const float* rt_g  = (const float*)d_in[28]; const float* rt_be = (const float*)d_in[29];
  const float* rt_W2 = (const float*)d_in[30]; const float* rt_b2 = (const float*)d_in[31];
  const float* Wp    = (const float*)d_in[32]; const float* bp    = (const float*)d_in[33];

  const int NT = in_sizes[0] / DD;
  const int NA = in_sizes[1] / DD;
  const int NB = in_sizes[2] / DD;
  const int E  = in_sizes[3];

  // Workspace layout
  float* bufAgg  = (float*)d_ws;                        // NT*DD
  float* bufZ    = bufAgg  + (size_t)NT * DD;           // NT*DD
  float* bufProj = bufZ    + (size_t)NT * DD;           // NT*DD
  unsigned short* wpk = (unsigned short*)(bufProj + (size_t)NT * DD); // 7*16384
  float* stats = (float*)(wpk + 7 * 16384);             // 256
  float* bnS   = stats + 256;                           // 128
  float* bnT   = bnS + DD;                              // 128

  float* out_hT = (float*)d_out;
  float* out_hA = out_hT + (size_t)NT * DD;
  float* out_hB = out_hA + (size_t)NA * DD;

  // Pack all weights to bf16 WMMA-B-fragment order
  pack_weight<<<64, 256, 0, stream>>>(ta_W1, wpk + 0 * 16384);
  pack_weight<<<64, 256, 0, stream>>>(ta_W2, wpk + 1 * 16384);
  pack_weight<<<64, 256, 0, stream>>>(tb_W1, wpk + 2 * 16384);
  pack_weight<<<64, 256, 0, stream>>>(tb_W2, wpk + 3 * 16384);
  pack_weight<<<64, 256, 0, stream>>>(rt_W1, wpk + 4 * 16384);
  pack_weight<<<64, 256, 0, stream>>>(rt_W2, wpk + 5 * 16384);
  pack_weight<<<64, 256, 0, stream>>>(Wp,    wpk + 6 * 16384);

  const int scatBlk = (E + 7) / 8;   // 8 edges (waves) per 256-thread block
  const int gA = (NA + 127) / 128;
  const int gB = (NB + 127) / 128;
  const int gT = (NT + 127) / 128;

  // ---- Phase A: aA = segsum(hT[src_ta] -> dst_ta); hA_new = MLP2((1+e)hA + aA)
  (void)hipMemsetAsync(bufAgg, 0, (size_t)NA * DD * sizeof(float), stream);
  scatter_add<<<scatBlk, 256, 0, stream>>>(hT, src_ta, dst_ta, bufAgg, E);
  (void)hipMemsetAsync(stats, 0, 256 * sizeof(float), stream);
  gemm128_t<true, false, true><<<gA, 256, 0, stream>>>(
      hA, bufAgg, eps_ta, nullptr, nullptr, wpk + 0 * 16384, ta_b1, bufZ, stats, NA);
  bn_make<<<1, DD, 0, stream>>>(stats, ta_g, ta_be, 1.0f / (float)NA, bnS, bnT);
  gemm128_t<false, true, false><<<gA, 256, 0, stream>>>(
      bufZ, nullptr, nullptr, bnS, bnT, wpk + 1 * 16384, ta_b2, out_hA, nullptr, NA);

  // ---- Phase B
  (void)hipMemsetAsync(bufAgg, 0, (size_t)NB * DD * sizeof(float), stream);
  scatter_add<<<scatBlk, 256, 0, stream>>>(hT, src_tb, dst_tb, bufAgg, E);
  (void)hipMemsetAsync(stats, 0, 256 * sizeof(float), stream);
  gemm128_t<true, false, true><<<gB, 256, 0, stream>>>(
      hB, bufAgg, eps_tb, nullptr, nullptr, wpk + 2 * 16384, tb_b1, bufZ, stats, NB);
  bn_make<<<1, DD, 0, stream>>>(stats, tb_g, tb_be, 1.0f / (float)NB, bnS, bnT);
  gemm128_t<false, true, false><<<gB, 256, 0, stream>>>(
      bufZ, nullptr, nullptr, bnS, bnT, wpk + 3 * 16384, tb_b2, out_hB, nullptr, NB);

  // ---- Phase T: hT_proj = hT@Wp+bp; aT = segsum(hA_new)+segsum(hB_new);
  //              hT_new = MLP2((1+e)hT_proj + aT)
  gemm128_t<false, false, false><<<gT, 256, 0, stream>>>(
      hT, nullptr, nullptr, nullptr, nullptr, wpk + 6 * 16384, bp, bufProj, nullptr, NT);
  (void)hipMemsetAsync(bufAgg, 0, (size_t)NT * DD * sizeof(float), stream);
  scatter_add<<<scatBlk, 256, 0, stream>>>(out_hA, src_at, dst_at, bufAgg, E);
  scatter_add<<<scatBlk, 256, 0, stream>>>(out_hB, src_bt, dst_bt, bufAgg, E);
  (void)hipMemsetAsync(stats, 0, 256 * sizeof(float), stream);
  gemm128_t<true, false, true><<<gT, 256, 0, stream>>>(
      bufProj, bufAgg, eps_rt, nullptr, nullptr, wpk + 4 * 16384, rt_b1, bufZ, stats, NT);
  bn_make<<<1, DD, 0, stream>>>(stats, rt_g, rt_be, 1.0f / (float)NT, bnS, bnT);
  gemm128_t<false, true, false><<<gT, 256, 0, stream>>>(
      bufZ, nullptr, nullptr, bnS, bnT, wpk + 5 * 16384, rt_b2, out_hT, nullptr, NT);
}
